// TDPN_SSFIB_HESSIAN_25744033972977
// MI455X (gfx1250) — matrix operand order repaired and checked by reading, not compiled, not run on hardware
//
#include <hip/hip_runtime.h>

// CDNA5 / gfx1250, wave32. Conv + MLP GEMMs via v_wmma_f32_16x16x32_f16.
// Activations kept as f16 NHWC; input tiles staged to LDS with
// GLOBAL_LOAD_ASYNC_TO_LDS_B128 (ASYNCcnt) on interior blocks.
typedef __attribute__((ext_vector_type(16))) _Float16 v16h;
typedef __attribute__((ext_vector_type(8)))  _Float16 v8h;
typedef __attribute__((ext_vector_type(8)))  float    v8f;

#define LDS_HALVES 24480   // max ks*TW*(Cin+8): 5*36*136

// ---------------------------------------------------------------------------
// Weight repack: OIHW f32 -> A-fragment-ordered f16.
//   wrep[(((mt*KSQ+pos)*CB+cb)*32 + lane)*16 + e], m=lane&15, half=lane>>4,
//   kk = e+16*half (contiguous k order, matches B), co=mt*16+m, ci=cb*32+kk
// ---------------------------------------------------------------------------
__global__ void repack_weights(const float* __restrict__ w, _Float16* __restrict__ wr,
                               int Cin, int ksq)
{
  long total = 64L * Cin * ksq;
  long idx = (long)blockIdx.x * 256 + threadIdx.x;
  if (idx >= total) return;
  int e    = (int)(idx & 15);
  int lane = (int)((idx >> 4) & 31);
  long t   = idx >> 9;
  int CB   = Cin >> 5;
  int cb   = (int)(t % CB); t /= CB;
  int pos  = (int)(t % ksq);
  int mt   = (int)(t / ksq);
  int m = lane & 15, half = lane >> 4;
  int kk = e + 16 * half;
  int co = mt * 16 + m, ci = cb * 32 + kk;
  wr[idx] = (_Float16)w[((size_t)co * Cin + ci) * ksq + pos];
}

// ---------------------------------------------------------------------------
// f32 NCHW input -> f16 NHWC
// ---------------------------------------------------------------------------
__global__ void cvt_nchw_to_nhwc(const float* __restrict__ x, _Float16* __restrict__ xh)
{
  long idx = (long)blockIdx.x * 256 + threadIdx.x;   // (b,h,w,c), c fastest
  int c = (int)(idx & 63); long t = idx >> 6;
  int w = (int)(t & 127); t >>= 7;
  int h = (int)(t & 127); int b = (int)(t >> 7);
  xh[idx] = (_Float16)x[(((size_t)b * 64 + c) * 128 + h) * 128 + w];
}

// ---------------------------------------------------------------------------
// Implicit-GEMM conv, NHWC f16 activations, Cout==64.
// Block = 128 threads (4 waves); wave wv owns out-channels [wv*16, wv*16+16).
// 2 N-tiles per wave -> 32 output pixels per block along W.
// K = Cin*ks*ks in 32-chunks (position-major). f32 accumulate in WMMA.
// ---------------------------------------------------------------------------
__global__ __launch_bounds__(128) void conv2d_wmma(
    const _Float16* __restrict__ in, int cin_tot,       // NHWC, Cin == cin_tot
    const _Float16* __restrict__ wrep,
    const float* __restrict__ bias,
    const _Float16* __restrict__ resid,                  // optional NHWC, 64 ch
    _Float16* __restrict__ out, int cout_tot, int cout_base,  // NHWC f16
    float* __restrict__ out32,                           // optional NCHW f32
    int Cin, int Hin, int Hout,
    int ks, int stride, int dil, int pad, int do_relu)
{
  __shared__ __align__(32) _Float16 sin_[LDS_HALVES];
  const int Win = Hin, Wout = Hout;
  const int KSQ = ks * ks;
  const int TW  = 31 * stride + (ks - 1) * dil + 1;   // staged width incl. halo
  const int RS  = Cin + 8;                             // LDS row stride (halves)
  const int b   = blockIdx.z;
  const int oh  = blockIdx.y;
  const int ow0 = blockIdx.x << 5;
  const int h0  = oh * stride - pad;
  const int w0  = ow0 * stride - pad;

  const bool interior = (h0 >= 0) && (h0 + (ks - 1) * dil < Hin) &&
                        (w0 >= 0) && (w0 + TW <= Win);
  if (interior) {
    // Pure byte-copy: async global->LDS, b128 per lane, tracked by ASYNCcnt.
    const int cpr = Cin >> 3;                 // 16B chunks per pixel-row
    const int nch = ks * TW * cpr;
    for (int i = threadIdx.x; i < nch; i += 128) {
      int r  = i / cpr, ch = i - r * cpr;
      int kh = r / TW,  x  = r - kh * TW;
      const _Float16* g = in +
          ((size_t)(b * Hin + h0 + kh * dil) * Win + (w0 + x)) * cin_tot + ch * 8;
      unsigned lv = (unsigned)(unsigned long long)&sin_[r * RS + ch * 8];
      asm volatile("global_load_async_to_lds_b128 %0, %1, off"
                   :: "v"(lv), "v"(g) : "memory");
    }
    asm volatile("s_wait_asynccnt 0x0" ::: "memory");
  } else {
    // Edge blocks: zero-padded VALU staging (coalesced: ci fastest both sides).
    for (int i = threadIdx.x; i < Cin * ks * TW; i += 128) {
      int ci = i % Cin; int r = i / Cin;
      int x = r % TW;   int kh = r / TW;
      int h = h0 + kh * dil, w = w0 + x;
      _Float16 v = (_Float16)0.f;
      if ((unsigned)h < (unsigned)Hin && (unsigned)w < (unsigned)Win)
        v = in[((size_t)(b * Hin + h) * Win + w) * cin_tot + ci];
      sin_[(kh * TW + x) * RS + ci] = v;
    }
  }
  __syncthreads();

  const int wv = threadIdx.x >> 5;
  const int lane = threadIdx.x & 31;
  const int n = lane & 15, half = lane >> 4;
  const int CB = Cin >> 5;
  v8f acc0 = {0.f,0.f,0.f,0.f,0.f,0.f,0.f,0.f};
  v8f acc1 = {0.f,0.f,0.f,0.f,0.f,0.f,0.f,0.f};
  const _Float16* wp = wrep + (size_t)wv * KSQ * CB * 512 + lane * 16;
  const int dN = 16 * stride * RS;

  for (int pos = 0; pos < KSQ; ++pos) {
    const int kh = pos / ks, kw = pos % ks;
    const int base0 = (kh * TW + n * stride + kw * dil) * RS + 16 * half;
    for (int cb = 0; cb < CB; ++cb) {
      v16h A  = *(const v16h*)(wp + (size_t)(pos * CB + cb) * 512);
      v16h B0 = *(const v16h*)&sin_[base0 + cb * 32];
      v16h B1 = *(const v16h*)&sin_[base0 + dN + cb * 32];
      acc0 = __builtin_amdgcn_wmma_f32_16x16x32_f16(false, A, false, B0,
                                                    (short)0, acc0, false, false);
      acc1 = __builtin_amdgcn_wmma_f32_16x16x32_f16(false, A, false, B1,
                                                    (short)0, acc1, false, false);
    }
  }

  const int cob = wv * 16 + 8 * half;   // 8 consecutive out-channels per lane
  #pragma unroll
  for (int t = 0; t < 2; ++t) {
    v8f a = t ? acc1 : acc0;
    int ow = ow0 + t * 16 + n;
    size_t pixbase = (size_t)(b * Hout + oh) * Wout + ow;
    float v[8];
    #pragma unroll
    for (int r = 0; r < 8; ++r) v[r] = a[r] + bias[cob + r];
    if (resid) {
      v8h rv = *(const v8h*)&resid[pixbase * 64 + cob];
      #pragma unroll
      for (int r = 0; r < 8; ++r) v[r] += (float)rv[r];
    }
    if (do_relu)
      #pragma unroll
      for (int r = 0; r < 8; ++r) v[r] = fmaxf(v[r], 0.f);
    if (out32) {
      #pragma unroll
      for (int r = 0; r < 8; ++r)
        out32[((size_t)(b * 64 + cob + r) * Hout + oh) * Wout + ow] = v[r];
    } else {
      v8h o;
      #pragma unroll
      for (int r = 0; r < 8; ++r) o[r] = (_Float16)v[r];
      *(v8h*)&out[pixbase * cout_tot + cout_base + cob] = o;   // 16B contiguous
    }
  }
}

// ---------------------------------------------------------------------------
// Unfold + mapping MLP (16 -> 256 -> 16) via two WMMA GEMMs.
// x1 is NHWC f16; x1m stays f32 [524288][16].
// ---------------------------------------------------------------------------
__global__ __launch_bounds__(128) void mapmlp_wmma(
    const _Float16* __restrict__ x1,  // [8][128][128][64]
    const float* __restrict__ w1,     // [256][16]
    const float* __restrict__ b1,     // [256]
    const float* __restrict__ w2,     // [16][256]
    const float* __restrict__ b2,     // [16]
    float* __restrict__ x1m)          // [524288][16]
{
  __shared__ _Float16 hbuf[4][16][256];   // 32 KB
  const int wv = threadIdx.x >> 5, lane = threadIdx.x & 31;
  const int n = lane & 15, half = lane >> 4;
  const long r0 = ((long)blockIdx.x * 4 + wv) * 16;

  long row = r0 + n;
  int c = (int)(row & 63); long t = row >> 6;
  int l = (int)(t & 1023); int b = (int)(t >> 10);
  int hk = l >> 5, wk = l & 31;
  const _Float16* xb = x1 + ((size_t)(b * 128 + hk * 4) * 128 + wk * 4) * 64 + c;
  v16h A;
  #pragma unroll
  for (int e = 0; e < 16; ++e) {
    int K = e + 16 * half;
    A[e] = (K < 16) ? xb[((K >> 2) * 128 + (K & 3)) * 64] : (_Float16)0.f;
  }

  for (int jt = 0; jt < 16; ++jt) {
    v16h Bf;
    #pragma unroll
    for (int e = 0; e < 16; ++e) {
      int k = e + 16 * half;
      Bf[e] = (_Float16)((k < 16) ? w1[(jt * 16 + n) * 16 + k] : 0.f);
    }
    v8f hc = {0.f,0.f,0.f,0.f,0.f,0.f,0.f,0.f};
    hc = __builtin_amdgcn_wmma_f32_16x16x32_f16(false, A, false, Bf,
                                                (short)0, hc, false, false);
    float bb = b1[jt * 16 + n];
    #pragma unroll
    for (int r = 0; r < 8; ++r)
      hbuf[wv][r + 8 * half][jt * 16 + n] = (_Float16)fmaxf(hc[r] + bb, 0.f);
  }
  __syncthreads();

  v8f acc = {0.f,0.f,0.f,0.f,0.f,0.f,0.f,0.f};
  for (int kc = 0; kc < 8; ++kc) {
    v16h A2, B2;
    #pragma unroll
    for (int e = 0; e < 16; ++e) {
      int kk = e + 16 * half;
      A2[e] = hbuf[wv][n][kc * 32 + kk];
      B2[e] = (_Float16)w2[n * 256 + kc * 32 + e + 16 * half];
    }
    acc = __builtin_amdgcn_wmma_f32_16x16x32_f16(false, A2, false, B2,
                                                 (short)0, acc, false, false);
  }
  float bb = b2[n];
  #pragma unroll
  for (int r = 0; r < 8; ++r)
    x1m[(r0 + r + 8 * half) * 16 + n] = acc[r] + bb;
}

// ---------------------------------------------------------------------------
// Keys cubic (a=-0.5) antialiased 8->4 dictionary downsample + L2 normalize.
// ---------------------------------------------------------------------------
__global__ void prep_dict(const float* __restrict__ dict, float* __restrict__ dn)
{
  int cls = threadIdx.x;  // 128
  float w[4][8];
  for (int i = 0; i < 4; ++i) {
    float cen = 2.f * i + 0.5f, s = 0.f;
    for (int x = 0; x < 8; ++x) {
      float tt = fabsf((x - cen) * 0.5f), wv;
      if (tt <= 1.f)      wv = ((1.5f * tt - 2.5f) * tt) * tt + 1.f;
      else if (tt < 2.f)  wv = (((-0.5f * tt + 2.5f) * tt) - 4.f) * tt + 2.f;
      else                wv = 0.f;
      w[i][x] = wv; s += wv;
    }
    for (int x = 0; x < 8; ++x) w[i][x] /= s;
  }
  const float* d = dict + cls * 64;
  float tmp[4][8];
  for (int i = 0; i < 4; ++i)
    for (int x = 0; x < 8; ++x) {
      float s = 0.f;
      for (int y = 0; y < 8; ++y) s += w[i][y] * d[y * 8 + x];
      tmp[i][x] = s;
    }
  float o[16]; float nrm = 0.f;
  for (int i = 0; i < 4; ++i)
    for (int j = 0; j < 4; ++j) {
      float s = 0.f;
      for (int x = 0; x < 8; ++x) s += w[j][x] * tmp[i][x];
      o[i * 4 + j] = s; nrm += s * s;
    }
  float inv = 1.f / fmaxf(sqrtf(nrm), 1e-8f);
  for (int p = 0; p < 16; ++p) dn[cls * 16 + p] = o[p] * inv;
}

// ---------------------------------------------------------------------------
// Cosine-sim argmax reconstruction + fold into cat2 (NHWC f16) channels 64..127.
// ---------------------------------------------------------------------------
__global__ __launch_bounds__(256) void recon_dict(
    const float* __restrict__ x1m, const float* __restrict__ dn,
    const float* __restrict__ dict, _Float16* __restrict__ cat2)
{
  __shared__ float sdn[2048];
  for (int i = threadIdx.x; i < 2048; i += 256) sdn[i] = dn[i];
  __syncthreads();
  long idx = (long)blockIdx.x * 256 + threadIdx.x;   // 0..524287
  const float* xr = x1m + idx * 16;
  float v[16], nrm = 0.f;
  for (int p = 0; p < 16; ++p) { v[p] = xr[p]; nrm += v[p] * v[p]; }
  float inv = 1.f / fmaxf(sqrtf(nrm), 1e-8f);
  float mx = -1e30f;
  for (int k = 0; k < 128; ++k) {
    float s = 0.f;
    for (int p = 0; p < 16; ++p) s += v[p] * sdn[k * 16 + p];
    mx = fmaxf(mx, s * inv);
  }
  float acc[64];
  for (int q = 0; q < 64; ++q) acc[q] = 0.f;
  for (int k = 0; k < 128; ++k) {
    float s = 0.f;
    for (int p = 0; p < 16; ++p) s += v[p] * sdn[k * 16 + p];
    s *= inv;
    if (s >= mx) {
      const float* dd = dict + k * 64;
      for (int q = 0; q < 64; ++q) acc[q] += s * dd[q];
    }
  }
  int c = (int)(idx & 63); long t = idx >> 6;
  int l = (int)(t & 1023); int b = (int)(t >> 10);
  int hk = l >> 5, wk = l & 31;
  for (int rr = 0; rr < 8; ++rr)
    for (int cc = 0; cc < 8; ++cc)
      cat2[((size_t)(b * 256 + hk * 8 + rr) * 256 + wk * 8 + cc) * 128 + 64 + c] =
          (_Float16)acc[rr * 8 + cc];
}

// ---------------------------------------------------------------------------
// Bicubic 2x upsample of fold(x1m) -> cat2 (NHWC f16) channels 0..63.
// Half-pixel sampling, edge-renormalized like jax.image.resize.
// ---------------------------------------------------------------------------
__global__ void upsample_bicubic(const float* __restrict__ x1m, _Float16* __restrict__ cat2)
{
  long idx = (long)blockIdx.x * 256 + threadIdx.x;   // (b,oh,ow,c), c fastest
  int c  = (int)(idx & 63); long t1 = idx >> 6;
  int ow = (int)(t1 & 255); long t2 = t1 >> 8;
  int oh = (int)(t2 & 255); int b = (int)(t2 >> 8);
  const float wA[4] = {-0.0234375f, 0.2265625f, 0.8671875f, -0.0703125f}; // frac .75
  const float wB[4] = {-0.0703125f, 0.8671875f, 0.2265625f, -0.0234375f}; // frac .25
  int ky = oh >> 1; int y0 = (oh & 1) ? ky - 1 : ky - 2; const float* wy = (oh & 1) ? wB : wA;
  int kx = ow >> 1; int x0 = (ow & 1) ? kx - 1 : kx - 2; const float* wx = (ow & 1) ? wB : wA;
  float sy = 0.f, sx = 0.f;
  for (int i = 0; i < 4; ++i) {
    if ((unsigned)(y0 + i) < 128u) sy += wy[i];
    if ((unsigned)(x0 + i) < 128u) sx += wx[i];
  }
  float s = 0.f;
  for (int i = 0; i < 4; ++i) {
    int y = y0 + i; if ((unsigned)y >= 128u) continue;
    float rs = 0.f;
    for (int j = 0; j < 4; ++j) {
      int x = x0 + j; if ((unsigned)x >= 128u) continue;
      int hk = y >> 2, wk = x >> 2, p = (y & 3) * 4 + (x & 3);
      rs += wx[j] * x1m[(((size_t)b * 1024 + hk * 32 + wk) * 64 + c) * 16 + p];
    }
    s += wy[i] * rs;
  }
  cat2[((size_t)(b * 256 + oh) * 256 + ow) * 128 + c] = (_Float16)(s / (sy * sx));
}

// ---------------------------------------------------------------------------
// attn = conv1x1(e, gate_w)+gate_b ; x3 *= attn  (NHWC f16, contiguous 64 ch)
// ---------------------------------------------------------------------------
__global__ void gate_mul(const _Float16* __restrict__ e, const float* __restrict__ gw,
                         const float* __restrict__ gb, _Float16* __restrict__ x3)
{
  long idx = (long)blockIdx.x * 256 + threadIdx.x;   // B*256*256 pixels
  size_t base = (size_t)idx * 64;
  float s = gb[0];
  for (int c = 0; c < 64; ++c) s += gw[c] * (float)e[base + c];
  for (int c = 0; c < 64; ++c)
    x3[base + c] = (_Float16)((float)x3[base + c] * s);
}

// ---------------------------------------------------------------------------
extern "C" void kernel_launch(void* const* d_in, const int* in_sizes, int n_in,
                              void* d_out, int out_size, void* d_ws, size_t ws_size,
                              hipStream_t stream)
{
  (void)in_sizes; (void)n_in; (void)out_size; (void)ws_size;
  // jax pytree (sorted-dict) flattening order
  enum { I_C1X1_B = 0, I_C1X1_W, I_DEC0_W, I_DEC0_B, I_DEC1_W, I_DEC1_B,
         I_DEC2_W, I_DEC2_B, I_DICT, I_ENC0_W, I_ENC0_B, I_ENC1_W, I_ENC1_B,
         I_ENC2_W, I_ENC2_B, I_GATE_B, I_GATE_W,
         I_M0 = 17, I_M1 = 27, I_M2 = 37,
         I_MAP_B1 = 47, I_MAP_B2, I_MAP_W1, I_MAP_W2, I_OUT_B, I_OUT_W, I_X = 53 };
  // within each m-block: +0 b11, +1 b31, +2 b32, +3 b51, +4 b52,
  //                      +5 w11, +6 w31, +7 w32, +8 w51, +9 w52
  auto fin = [&](int i) { return (const float*)d_in[i]; };

  char* wsb = (char*)d_ws; size_t off = 0;
  auto alloc = [&](size_t bytes) -> char* {
    char* p = wsb + off; off += (bytes + 255) & ~(size_t)255; return p;
  };
  _Float16* wr   = (_Float16*)alloc(4ull << 20);
  float*    dn   = (float*)alloc(2048 * 4);
  float*    x1m  = (float*)alloc((size_t)524288 * 16 * 4);
  _Float16* xh   = (_Float16*)alloc((size_t)8 * 128 * 128 * 64 * 2);
  _Float16* cc1  = (_Float16*)alloc((size_t)8 * 128 * 128 * 128 * 2);
  _Float16* cc2  = (_Float16*)alloc((size_t)8 * 128 * 128 * 128 * 2);
  _Float16* bufA = (_Float16*)alloc((size_t)8 * 128 * 128 * 64 * 2);
  _Float16* bufB = (_Float16*)alloc((size_t)8 * 128 * 128 * 64 * 2);
  _Float16* cat2 = (_Float16*)alloc((size_t)8 * 256 * 256 * 128 * 2);
  _Float16* x3   = (_Float16*)alloc((size_t)8 * 256 * 256 * 64 * 2);
  _Float16* e1   = (_Float16*)alloc((size_t)8 * 256 * 256 * 64 * 2);
  _Float16* e2   = (_Float16*)alloc((size_t)8 * 256 * 256 * 64 * 2);

  size_t woff = 0;
  auto rep = [&](int widx, int Cin, int ksq) -> const _Float16* {
    _Float16* dst = wr + woff;
    long total = 64L * Cin * ksq; woff += (size_t)total;
    repack_weights<<<dim3((unsigned)((total + 255) / 256)), dim3(256), 0, stream>>>(
        fin(widx), dst, Cin, ksq);
    return dst;
  };
  auto conv = [&](const _Float16* in, int cin_tot, const _Float16* w,
                  const float* bias, const _Float16* resid,
                  _Float16* out, int cout_tot, int cout_base, float* out32,
                  int Cin, int Hin, int Hout,
                  int ks, int stride, int dil, int pad, int relu) {
    dim3 grid((unsigned)(Hout / 32), (unsigned)Hout, 8);
    conv2d_wmma<<<grid, 128, 0, stream>>>(in, cin_tot, w, bias, resid,
                                          out, cout_tot, cout_base, out32,
                                          Cin, Hin, Hout, ks, stride, dil, pad, relu);
  };

  // Repack all conv weights into A-fragment layout (deterministic, per call).
  const _Float16 *mw31[3], *mw51[3], *mw32[3], *mw52[3], *mw11[3];
  const int mbase[3] = { I_M0, I_M1, I_M2 };
  for (int i = 0; i < 3; ++i) {
    mw31[i] = rep(mbase[i] + 6,  64,  9);
    mw51[i] = rep(mbase[i] + 8,  64, 25);
    mw32[i] = rep(mbase[i] + 7, 128,  9);
    mw52[i] = rep(mbase[i] + 9, 128, 25);
    mw11[i] = rep(mbase[i] + 5, 128,  1);
  }
  const _Float16* c11  = rep(I_C1X1_W, 128, 1);
  const _Float16* we[3] = { rep(I_ENC0_W, 64, 9), rep(I_ENC1_W, 64, 9), rep(I_ENC2_W, 64, 9) };
  const _Float16* wd[3] = { rep(I_DEC0_W, 64, 9), rep(I_DEC1_W, 64, 9), rep(I_DEC2_W, 64, 9) };
  const _Float16* wout = rep(I_OUT_W, 64, 9);

  cvt_nchw_to_nhwc<<<32768, 256, 0, stream>>>(fin(I_X), xh);
  prep_dict<<<1, 128, 0, stream>>>(fin(I_DICT), dn);

  // 3 x MSRB at 128x128 (all f16 NHWC)
  auto msrb = [&](const _Float16* X, int mi, int i, _Float16* Y) {
    conv(X,   64,  mw31[i], fin(mi + 1), nullptr, cc1, 128,  0, nullptr,  64, 128, 128, 3, 1, 1, 1, 1);
    conv(X,   64,  mw51[i], fin(mi + 3), nullptr, cc1, 128, 64, nullptr,  64, 128, 128, 5, 1, 1, 2, 1);
    conv(cc1, 128, mw32[i], fin(mi + 2), nullptr, cc2, 128,  0, nullptr, 128, 128, 128, 3, 1, 1, 1, 1);
    conv(cc1, 128, mw52[i], fin(mi + 4), nullptr, cc2, 128, 64, nullptr, 128, 128, 128, 5, 1, 1, 2, 1);
    conv(cc2, 128, mw11[i], fin(mi + 0), X,       Y,    64,  0, nullptr, 128, 128, 128, 1, 1, 1, 0, 0);
  };
  msrb(xh,   I_M0, 0, bufA);
  msrb(bufA, I_M1, 1, bufB);
  msrb(bufB, I_M2, 2, bufA);     // x1 = bufA

  // Unfold + mapping MLP (WMMA), dictionary recon + fold, bicubic 2x upsample
  mapmlp_wmma<<<8192, 128, 0, stream>>>(bufA, fin(I_MAP_W1), fin(I_MAP_B1),
                                        fin(I_MAP_W2), fin(I_MAP_B2), x1m);
  recon_dict<<<2048, 256, 0, stream>>>(x1m, dn, fin(I_DICT), cat2);
  upsample_bicubic<<<131072, 256, 0, stream>>>(x1m, cat2);

  // conv1x1 over concat(x1u, x2f) -> x3 @ 256x256
  conv(cat2, 128, c11, fin(I_C1X1_B), nullptr, x3, 64, 0, nullptr, 128, 256, 256, 1, 1, 1, 0, 0);

  // encoder / decoder dilated convs (dil 1,2,4,4,2,1), all ReLU
  conv(x3, 64, we[0], fin(I_ENC0_B), nullptr, e1, 64, 0, nullptr, 64, 256, 256, 3, 1, 1, 1, 1);
  conv(e1, 64, we[1], fin(I_ENC1_B), nullptr, e2, 64, 0, nullptr, 64, 256, 256, 3, 1, 2, 2, 1);
  conv(e2, 64, we[2], fin(I_ENC2_B), nullptr, e1, 64, 0, nullptr, 64, 256, 256, 3, 1, 4, 4, 1);
  conv(e1, 64, wd[0], fin(I_DEC0_B), nullptr, e2, 64, 0, nullptr, 64, 256, 256, 3, 1, 4, 4, 1);
  conv(e2, 64, wd[1], fin(I_DEC1_B), nullptr, e1, 64, 0, nullptr, 64, 256, 256, 3, 1, 2, 2, 1);
  conv(e1, 64, wd[2], fin(I_DEC2_B), nullptr, e2, 64, 0, nullptr, 64, 256, 256, 3, 1, 1, 1, 1);

  // gating, then final stride-2 conv into d_out ([8,64,128,128] f32 NCHW)
  gate_mul<<<2048, 256, 0, stream>>>(e2, fin(I_GATE_W), fin(I_GATE_B), x3);
  conv(x3, 64, wout, fin(I_OUT_B), nullptr, nullptr, 0, 0, (float*)d_out,
       64, 256, 128, 3, 2, 1, 1, 0);
}
